// SparseAttention_72353019068580
// MI455X (gfx1250) — matrix operand order repaired and checked by reading, not compiled, not run on hardware
//
#include <hip/hip_runtime.h>
#include <hip/hip_bf16.h>

// ---------------------------------------------------------------------------
// Types for WMMA fragments (gfx1250, wave32)
// ---------------------------------------------------------------------------
typedef _Float16 h8   __attribute__((ext_vector_type(8)));
typedef _Float16 v16h __attribute__((ext_vector_type(16)));
typedef float    v8f  __attribute__((ext_vector_type(8)));

#define HH   16      // query heads
#define KVH  8
#define GQ   2
#define NSEQ 2048
#define DIM  1024
#define DD   64
#define WS   64
#define CBS  32
#define STRIDE 16
#define NCB  128
#define NFB  64
#define NSEL 4
#define CK   129     // NMEM + NCB
#define CKP  160     // CK padded to multiple of 32
#define HID  2048
#define SCALE 0.125f
#define NEGBIG  -3.402823466e38f
#define NEG10   -3.402823466e37f

// ---------------------------------------------------------------------------
// Batched register-blocked WMMA GEMM with cross-iteration double buffering:
//   C[z] = alpha * A[z] @ Bt[z/bdiv]^T + bias (opt ReLU, opt f16 copy)
//   A  : M x K row-major f16; Bt : N x K row-major f16 (B transposed)
// Wave computes a 64x32 output block = 4x2 WMMA tiles (8 acc, 8 WMMA / k-step).
// Fragments for k-step s+1 are loaded while the 8 WMMAs of step s execute
// (ping-pong register buffers, no steady-state branch, no register rotation).
// Out-of-range fragment rows/cols are CLAMPED for loads (stays in-bounds;
// affected output rows/cols are discarded by the guarded store).
// K must be a multiple of 32.
// ---------------------------------------------------------------------------
__global__ __launch_bounds__(128)
void wmma_gemm_kernel(const _Float16* __restrict__ A,
                      const _Float16* __restrict__ Bt,
                      float* __restrict__ C,
                      _Float16* __restrict__ Ch,
                      const float* __restrict__ bias,
                      int M, int N, int K,
                      float alpha, int relu,
                      long long strideA, long long strideB, long long strideC,
                      int bdivB)
{
    const int lane    = threadIdx.x & 31;
    const int wave    = threadIdx.x >> 5;
    const int half_id = lane >> 4;
    const int lane16  = lane & 15;
    const int z       = blockIdx.z;

    const _Float16* Ab = A  + (long long)z * strideA;
    const _Float16* Bb = Bt + (long long)(z / bdivB) * strideB;

    const int rowBase = blockIdx.y * 64;
    const int colBase = (blockIdx.x * 4 + wave) * 32;

    // Clamped fragment row pointers (A: +half_id*16 pre-folded; B: +half_id*16)
    const _Float16* arow[4];
    const _Float16* brow[2];
#pragma unroll
    for (int mi = 0; mi < 4; ++mi) {
        int m = rowBase + mi * 16 + lane16;
        if (m > M - 1) m = M - 1;
        arow[mi] = Ab + (long long)m * K + half_id * 16;
    }
#pragma unroll
    for (int ni = 0; ni < 2; ++ni) {
        int n = colBase + ni * 16 + lane16;
        if (n > N - 1) n = N - 1;
        brow[ni] = Bb + (long long)n * K + half_id * 16;
    }

    v8f acc[4][2];
#pragma unroll
    for (int mi = 0; mi < 4; ++mi)
#pragma unroll
        for (int ni = 0; ni < 2; ++ni)
            acc[mi][ni] = (v8f){0.f, 0.f, 0.f, 0.f, 0.f, 0.f, 0.f, 0.f};

    // A 16x32 f16 layout: lanes 0-15 hold K[0..15],
    //                     lanes 16-31 hold K[16..31]
    auto loadA = [&](v16h (&dst)[4], int kk) {
#pragma unroll
        for (int mi = 0; mi < 4; ++mi) {
            const h8 lo = *(const h8*)(arow[mi] + kk);
            const h8 hi = *(const h8*)(arow[mi] + kk + 8);
#pragma unroll
            for (int i = 0; i < 8; ++i) { dst[mi][i] = lo[i]; dst[mi][8 + i] = hi[i]; }
        }
    };
    // B 32x16 f16 layout: lanes 0-15 hold K[0..15], lanes 16-31 K[16..31]
    auto loadB = [&](v16h (&dst)[2], int kk) {
#pragma unroll
        for (int ni = 0; ni < 2; ++ni) {
            const h8 lo = *(const h8*)(brow[ni] + kk);
            const h8 hi = *(const h8*)(brow[ni] + kk + 8);
#pragma unroll
            for (int i = 0; i < 8; ++i) { dst[ni][i] = lo[i]; dst[ni][8 + i] = hi[i]; }
        }
    };
    auto mac = [&](v16h (&av)[4], v16h (&bv)[2]) {
#pragma unroll
        for (int mi = 0; mi < 4; ++mi)
#pragma unroll
            for (int ni = 0; ni < 2; ++ni)
                acc[mi][ni] = __builtin_amdgcn_wmma_f32_16x16x32_f16(
                    false, av[mi], false, bv[ni], (short)0, acc[mi][ni],
                    false, false);
    };

    const int S = K >> 5;         // number of 32-wide k-steps (>= 1)
    v16h a0[4], b0[2], a1[4], b1[2];
    loadA(a0, 0); loadB(b0, 0);
    int s = 0;
    for (; s + 2 < S; s += 2) {
        loadA(a1, (s + 1) << 5); loadB(b1, (s + 1) << 5);
        mac(a0, b0);
        loadA(a0, (s + 2) << 5); loadB(b0, (s + 2) << 5);
        mac(a1, b1);
    }
    if (s + 1 < S) {
        loadA(a1, (s + 1) << 5); loadB(b1, (s + 1) << 5);
        mac(a0, b0);
        mac(a1, b1);
    } else {
        mac(a0, b0);
    }

    float*     Cb  = C  ? C  + (long long)z * strideC : nullptr;
    _Float16*  Chb = Ch ? Ch + (long long)z * strideC : nullptr;
#pragma unroll
    for (int ni = 0; ni < 2; ++ni) {
        const int nn = colBase + ni * 16 + lane16;
        if (nn >= N) continue;
        const float bval = bias ? bias[nn] : 0.f;
#pragma unroll
        for (int mi = 0; mi < 4; ++mi) {
#pragma unroll
            for (int r = 0; r < 8; ++r) {
                const int mm = rowBase + mi * 16 + r + half_id * 8;
                if (mm >= M) continue;
                float val = acc[mi][ni][r] * alpha + bval;
                if (relu) val = val > 0.f ? val : 0.f;
                const long long offo = (long long)mm * N + nn;
                if (Cb)  Cb[offo]  = val;
                if (Chb) Chb[offo] = (_Float16)val;
            }
        }
    }
}

// ---------------------------------------------------------------------------
// Convert f32 weight (K x N) -> f16 transposed (N x K)
// ---------------------------------------------------------------------------
__global__ void convT_kernel(const float* __restrict__ W, _Float16* __restrict__ Wt,
                             int K, int N)
{
    long long i = (long long)blockIdx.x * 256 + threadIdx.x;
    if (i >= (long long)K * N) return;
    int       n = (int)(i % N);
    long long k = i / N;
    Wt[(long long)n * K + k] = (_Float16)W[i];
}

// ---------------------------------------------------------------------------
// RMSNorm: xh = f16( inp * rsqrt(mean(inp^2)+eps) * g )
// ---------------------------------------------------------------------------
__global__ __launch_bounds__(256)
void rmsnorm_kernel(const float* __restrict__ inp, const float* __restrict__ g,
                    _Float16* __restrict__ xh)
{
    const int n = blockIdx.x;
    const float* row = inp + (long long)n * DIM;
    __shared__ float red[256];
    float s = 0.f;
    for (int c = threadIdx.x; c < DIM; c += 256) { float v = row[c]; s += v * v; }
    red[threadIdx.x] = s;
    __syncthreads();
    for (int o = 128; o > 0; o >>= 1) {
        if (threadIdx.x < o) red[threadIdx.x] += red[threadIdx.x + o];
        __syncthreads();
    }
    const float r = rsqrtf(red[0] / (float)DIM + 1.1920929e-7f);
    _Float16* orow = xh + (long long)n * DIM;
    for (int c = threadIdx.x; c < DIM; c += 256)
        orow[c] = (_Float16)(row[c] * r * g[c]);
}

// ---------------------------------------------------------------------------
// Split qkv (N x 2048) into head-major q/k/v f32 (+ q f16 for WMMA)
// ---------------------------------------------------------------------------
__global__ void split_qkv_kernel(const float* __restrict__ qkv,
                                 float* __restrict__ q, float* __restrict__ k,
                                 float* __restrict__ v, _Float16* __restrict__ qh)
{
    long long i = (long long)blockIdx.x * 256 + threadIdx.x;
    if (i >= (long long)NSEQ * 2048) return;
    const int c = (int)(i & 2047);
    const int n = (int)(i >> 11);
    const float val = qkv[i];
    if (c < HH * DD) {
        const int h = c >> 6, d = c & 63;
        const long long o = ((long long)h * NSEQ + n) * DD + d;
        q[o] = val; qh[o] = (_Float16)val;
    } else if (c < HH * DD + KVH * DD) {
        const int hc = c - HH * DD;
        k[((long long)(hc >> 6) * NSEQ + n) * DD + (hc & 63)] = val;
    } else {
        const int hc = c - HH * DD - KVH * DD;
        v[((long long)(hc >> 6) * NSEQ + n) * DD + (hc & 63)] = val;
    }
}

// ---------------------------------------------------------------------------
// Build compression-window inputs kw/vw as f16 A matrices (rows kvh*128+cb, K=2048)
// ---------------------------------------------------------------------------
__global__ void window_kernel(const float* __restrict__ k, const float* __restrict__ v,
                              const float* __restrict__ k_pos, const float* __restrict__ v_pos,
                              _Float16* __restrict__ kwA, _Float16* __restrict__ vwA)
{
    long long i = (long long)blockIdx.x * 256 + threadIdx.x;   // KVH*NCB*CBS*D
    if (i >= (long long)KVH * NCB * CBS * DD) return;
    const int d  = (int)(i & 63);
    long long r  = i >> 6;
    const int t  = (int)(r & (CBS - 1)); r >>= 5;
    const int cb = (int)(r & (NCB - 1));
    const int kvh = (int)(r >> 7);
    const int pos = cb * STRIDE - (CBS - STRIDE) + t;
    float kv = 0.f, vv = 0.f;
    if (pos >= 0) {
        const long long src = ((long long)kvh * NSEQ + pos) * DD + d;
        kv = k[src]; vv = v[src];
    }
    const long long pidx = ((long long)kvh * CBS + t) * DD + d;
    const long long o = (((long long)kvh * NCB + cb) * CBS + t) * DD + d;
    kwA[o] = (_Float16)(kv + k_pos[pidx]);
    vwA[o] = (_Float16)(vv + v_pos[pidx]);
}

// ---------------------------------------------------------------------------
// Assemble ck (f16, 8x129x64), cv (f32) and cvT (f16, 8x64x160, pre-zeroed)
// ---------------------------------------------------------------------------
__global__ void assemble_c_kernel(const float* __restrict__ ckbody,
                                  const float* __restrict__ cvbody,
                                  const float* __restrict__ mem_kv,
                                  _Float16* __restrict__ ck_h,
                                  float* __restrict__ cv_f,
                                  _Float16* __restrict__ cvT_h)
{
    long long i = (long long)blockIdx.x * 256 + threadIdx.x;  // KVH*129*64
    if (i >= (long long)KVH * CK * DD) return;
    const int d = (int)(i & 63);
    long long r = i >> 6;
    const int j   = (int)(r % CK);
    const int kvh = (int)(r / CK);
    float kvv, vvv;
    if (j == 0) {
        kvv = mem_kv[((long long)0 * KVH + kvh) * DD + d];  // mem_kv[0][kvh][0][d]
        vvv = mem_kv[((long long)1 * KVH + kvh) * DD + d];
    } else {
        const long long b = ((long long)kvh * NCB + (j - 1)) * DD + d;
        kvv = ckbody[b]; vvv = cvbody[b];
    }
    ck_h[i] = (_Float16)kvv;
    cv_f[i] = vvv;
    cvT_h[((long long)kvh * DD + d) * CKP + j] = (_Float16)vvv;
}

// ---------------------------------------------------------------------------
// Softmax over 129 compressed keys -> f16 attn (padded to 160 with zeros)
// One wave per (h, n).
// ---------------------------------------------------------------------------
__global__ __launch_bounds__(128)
void csoftmax_kernel(const float* __restrict__ csim, _Float16* __restrict__ cattn_h)
{
    const int w    = blockIdx.x * 4 + (threadIdx.x >> 5);
    const int lane = threadIdx.x & 31;
    const int h = w >> 11, n = w & (NSEQ - 1);
    const float* row = csim + ((long long)h * NSEQ + n) * CK;
    float vals[5];
    float mx = NEGBIG;
#pragma unroll
    for (int t = 0; t < 5; ++t) {
        const int j = lane + 32 * t;
        const float v = (j < CK) ? row[j] : NEGBIG;
        vals[t] = v; mx = fmaxf(mx, v);
    }
    for (int o = 16; o > 0; o >>= 1) mx = fmaxf(mx, __shfl_xor(mx, o, 32));
    float s = 0.f;
#pragma unroll
    for (int t = 0; t < 5; ++t) {
        const int j = lane + 32 * t;
        const float e = (j < CK) ? __expf(vals[t] - mx) : 0.f;
        vals[t] = e; s += e;
    }
    for (int o = 16; o > 0; o >>= 1) s += __shfl_xor(s, o, 32);
    const float inv = 1.f / s;
    _Float16* orow = cattn_h + ((long long)h * NSEQ + n) * CKP;
#pragma unroll
    for (int t = 0; t < 5; ++t) {
        const int j = lane + 32 * t;
        if (j < CKP) orow[j] = (_Float16)((j < CK) ? vals[t] * inv : 0.f);
    }
}

// ---------------------------------------------------------------------------
// Interleaved RoPE for q (heads 0..15) and k (heads 16..23)
// ---------------------------------------------------------------------------
__global__ void rope_kernel(const float* __restrict__ q, const float* __restrict__ k,
                            float* __restrict__ rq, float* __restrict__ rk)
{
    long long i = (long long)blockIdx.x * 256 + threadIdx.x;  // (16+8)*2048*32
    if (i >= (long long)(HH + KVH) * NSEQ * (DD / 2)) return;
    const int p = (int)(i & 31);
    long long r = i >> 5;
    const int n  = (int)(r & (NSEQ - 1));
    const int hh = (int)(r >> 11);
    const float inv_freq = __expf(-((float)(2 * p) / (float)DD) * 9.210340372f); // ln(1e4)
    const float ang = (float)n * inv_freq;
    const float c = __cosf(ang), s = __sinf(ang);
    const float* src; float* dst;
    if (hh < HH) {
        src = q  + ((long long)hh * NSEQ + n) * DD;
        dst = rq + ((long long)hh * NSEQ + n) * DD;
    } else {
        src = k  + ((long long)(hh - HH) * NSEQ + n) * DD;
        dst = rk + ((long long)(hh - HH) * NSEQ + n) * DD;
    }
    const float t0 = src[2 * p], t1 = src[2 * p + 1];
    dst[2 * p]     = t0 * c - t1 * s;
    dst[2 * p + 1] = t1 * c + t0 * s;
}

// ---------------------------------------------------------------------------
// Importance scores + softmax(+pad logit) + top-4 selection. Thread per (kvh,n)
// ---------------------------------------------------------------------------
__global__ void topk_kernel(const float* __restrict__ csim,
                            int* __restrict__ selidx, int* __restrict__ selok)
{
    const int i = blockIdx.x * 256 + threadIdx.x;   // KVH*NSEQ
    if (i >= KVH * NSEQ) return;
    const int n = i & (NSEQ - 1);
    const int kvh = i >> 11;
    const float* r0 = csim + ((long long)(kvh * GQ) * NSEQ + n) * CK;
    const float* r1 = r0 + (long long)NSEQ * CK;
    float m[NFB];
    const int own = n >> 5;
    float mx = -1000.f;   // pad logit
#pragma unroll 4
    for (int fb = 0; fb < NFB; ++fb) {
        float v = 0.25f * (r0[1 + 2 * fb] + r0[2 + 2 * fb] +
                           r1[1 + 2 * fb] + r1[2 + 2 * fb]);
        if (fb == own) v = NEGBIG;
        m[fb] = v; mx = fmaxf(mx, v);
    }
    float s = __expf(-1000.f - mx);
#pragma unroll 4
    for (int fb = 0; fb < NFB; ++fb) { m[fb] = __expf(m[fb] - mx); s += m[fb]; }
    const float inv = 1.f / s;
#pragma unroll
    for (int t = 0; t < NSEL; ++t) {
        float best = -1.f; int bi = 0;
        for (int fb = 0; fb < NFB; ++fb)
            if (m[fb] > best) { best = m[fb]; bi = fb; }
        selidx[i * NSEL + t] = bi;
        selok[i * NSEL + t]  = (best * inv > 1e-10f) ? 1 : 0;
        m[bi] = -2.f;
    }
}

// ---------------------------------------------------------------------------
// Fine (selected-block) attention. One wave per (kvh, n); both GQ heads.
// 5 blocks x 32 keys; slot 4 = own block (causal). float4 vectorized dots.
// ---------------------------------------------------------------------------
__global__ __launch_bounds__(128)
void fine_attn_kernel(const float* __restrict__ rq, const float* __restrict__ rk,
                      const float* __restrict__ v,
                      const int* __restrict__ selidx, const int* __restrict__ selok,
                      float* __restrict__ fout)
{
    __shared__ float probs[4][2][160];
    const int wave = threadIdx.x >> 5, lane = threadIdx.x & 31;
    const int w = blockIdx.x * 4 + wave;
    const int n = w & (NSEQ - 1), kvh = w >> 11;
    int blk[5], okm[5];
#pragma unroll
    for (int t = 0; t < NSEL; ++t) {
        blk[t] = selidx[(w << 2) + t];
        okm[t] = selok[(w << 2) + t];
    }
    blk[4] = n >> 5; okm[4] = 1;
    const int qi = n & 31;
    const float4* q0 = (const float4*)(rq + ((long long)(kvh * GQ) * NSEQ + n) * DD);
    const float4* q1 = q0 + (long long)NSEQ * (DD / 4);

    float sims[5][2];
#pragma unroll
    for (int t = 0; t < 5; ++t) {
        const int pos = blk[t] * 32 + lane;
        const bool valid = (t < 4) ? (okm[t] != 0) : (lane <= qi);
        const float4* kr = (const float4*)(rk + ((long long)kvh * NSEQ + pos) * DD);
        float s0 = 0.f, s1 = 0.f;
#pragma unroll
        for (int d = 0; d < DD / 4; ++d) {
            const float4 kv = kr[d];
            const float4 a0 = q0[d], a1 = q1[d];
            s0 += a0.x * kv.x + a0.y * kv.y + a0.z * kv.z + a0.w * kv.w;
            s1 += a1.x * kv.x + a1.y * kv.y + a1.z * kv.z + a1.w * kv.w;
        }
        sims[t][0] = valid ? s0 * SCALE : NEG10;
        sims[t][1] = valid ? s1 * SCALE : NEG10;
    }
#pragma unroll
    for (int g = 0; g < 2; ++g) {
        float mx = NEGBIG;
#pragma unroll
        for (int t = 0; t < 5; ++t) mx = fmaxf(mx, sims[t][g]);
        for (int o = 16; o > 0; o >>= 1) mx = fmaxf(mx, __shfl_xor(mx, o, 32));
        float s = 0.f;
#pragma unroll
        for (int t = 0; t < 5; ++t) { const float e = __expf(sims[t][g] - mx); sims[t][g] = e; s += e; }
        for (int o = 16; o > 0; o >>= 1) s += __shfl_xor(s, o, 32);
        const float inv = 1.f / s;
#pragma unroll
        for (int t = 0; t < 5; ++t) probs[wave][g][t * 32 + lane] = sims[t][g] * inv;
    }
    __syncthreads();
    // lane covers dims 2*lane, 2*lane+1
    float a00 = 0, a01 = 0, a10 = 0, a11 = 0;
#pragma unroll
    for (int t = 0; t < 5; ++t) {
        const int base = blk[t] * 32;
        for (int j = 0; j < 32; ++j) {
            const float2 vv = *(const float2*)(v + ((long long)kvh * NSEQ + base + j) * DD + 2 * lane);
            const float p0 = probs[wave][0][t * 32 + j];
            const float p1 = probs[wave][1][t * 32 + j];
            a00 += p0 * vv.x; a01 += p0 * vv.y;
            a10 += p1 * vv.x; a11 += p1 * vv.y;
        }
    }
    float* o0 = fout + ((long long)(kvh * GQ) * NSEQ + n) * DD;
    float* o1 = o0 + (long long)NSEQ * DD;
    o0[2 * lane] = a00; o0[2 * lane + 1] = a01;
    o1[2 * lane] = a10; o1[2 * lane + 1] = a11;
}

// ---------------------------------------------------------------------------
// Sliding-window attention: key positions [n-63, n]. One wave per (h, n).
// ---------------------------------------------------------------------------
__global__ __launch_bounds__(128)
void sliding_kernel(const float* __restrict__ rq, const float* __restrict__ rk,
                    const float* __restrict__ v, float* __restrict__ sout)
{
    __shared__ float probs[4][64];
    const int wave = threadIdx.x >> 5, lane = threadIdx.x & 31;
    const int w = blockIdx.x * 4 + wave;
    const int n = w & (NSEQ - 1), h = w >> 11;
    const int kvh = h >> 1;
    const float4* q = (const float4*)(rq + ((long long)h * NSEQ + n) * DD);
    float sims[2];
#pragma unroll
    for (int t = 0; t < 2; ++t) {
        const int j = lane + 32 * t;
        const int p = n - (WS - 1) + j;
        if (p >= 0) {
            const float4* kr = (const float4*)(rk + ((long long)kvh * NSEQ + p) * DD);
            float s = 0.f;
#pragma unroll
            for (int d = 0; d < DD / 4; ++d) {
                const float4 a = q[d], b = kr[d];
                s += a.x * b.x + a.y * b.y + a.z * b.z + a.w * b.w;
            }
            sims[t] = s * SCALE;
        } else sims[t] = NEGBIG;
    }
    float mx = fmaxf(sims[0], sims[1]);
    for (int o = 16; o > 0; o >>= 1) mx = fmaxf(mx, __shfl_xor(mx, o, 32));
    float s = 0.f;
#pragma unroll
    for (int t = 0; t < 2; ++t) { const float e = __expf(sims[t] - mx); sims[t] = e; s += e; }
    for (int o = 16; o > 0; o >>= 1) s += __shfl_xor(s, o, 32);
    const float inv = 1.f / s;
    probs[wave][lane]      = sims[0] * inv;
    probs[wave][lane + 32] = sims[1] * inv;
    __syncthreads();
    float a0 = 0, a1 = 0;
    for (int j = 0; j < WS; ++j) {
        const int p = n - (WS - 1) + j;
        if (p < 0) continue;
        const float2 vv = *(const float2*)(v + ((long long)kvh * NSEQ + p) * DD + 2 * lane);
        const float pp = probs[wave][j];
        a0 += pp * vv.x; a1 += pp * vv.y;
    }
    float* o = sout + ((long long)h * NSEQ + n) * DD;
    o[2 * lane] = a0; o[2 * lane + 1] = a1;
}

// ---------------------------------------------------------------------------
// Gated combine -> f16 rows (N x H*D) for final projection GEMM
// ---------------------------------------------------------------------------
__global__ void combine_kernel(const float* __restrict__ gatesRaw,
                               const float* __restrict__ cout_,
                               const float* __restrict__ fout,
                               const float* __restrict__ sout,
                               _Float16* __restrict__ comb_h)
{
    long long i = (long long)blockIdx.x * 256 + threadIdx.x;  // NSEQ*DIM
    if (i >= (long long)NSEQ * DIM) return;
    const int c = (int)(i & (DIM - 1));
    const int n = (int)(i >> 10);
    const int h = c >> 6, d = c & 63;
    const float* gr = gatesRaw + (long long)n * (3 * HH) + h * 3;
    const float g0 = 1.f / (1.f + __expf(-gr[0]));
    const float g1 = 1.f / (1.f + __expf(-gr[1]));
    const float g2 = 1.f / (1.f + __expf(-gr[2]));
    const long long hn = ((long long)h * NSEQ + n) * DD + d;
    comb_h[i] = (_Float16)(g0 * cout_[hn] + g1 * fout[hn] + g2 * sout[hn]);
}

// ---------------------------------------------------------------------------
// Host-side launch
// ---------------------------------------------------------------------------
static inline size_t align256(size_t x) { return (x + 255) & ~(size_t)255; }

static inline void launch_gemm(const _Float16* A, const _Float16* Bt,
                               float* C, _Float16* Ch, const float* bias,
                               int M, int N, int K, float alpha, int relu,
                               long long sA, long long sB, long long sC,
                               int bdiv, int batch, hipStream_t stream)
{
    dim3 grid((N + 127) / 128, (M + 63) / 64, batch);
    wmma_gemm_kernel<<<grid, 128, 0, stream>>>(A, Bt, C, Ch, bias, M, N, K,
                                               alpha, relu, sA, sB, sC, bdiv);
}

extern "C" void kernel_launch(void* const* d_in, const int* in_sizes, int n_in,
                              void* d_out, int out_size, void* d_ws, size_t ws_size,
                              hipStream_t stream)
{
    (void)in_sizes; (void)n_in; (void)out_size; (void)ws_size;
    const float* inp    = (const float*)d_in[0];
    const float* norm_g = (const float*)d_in[1];
    const float* Wqkv   = (const float*)d_in[2];
    const float* mem_kv = (const float*)d_in[3];
    const float* k_pos  = (const float*)d_in[4];
    const float* v_pos  = (const float*)d_in[5];
    const float* kW1    = (const float*)d_in[6];
    const float* kb1    = (const float*)d_in[7];
    const float* kW2    = (const float*)d_in[8];
    const float* kb2    = (const float*)d_in[9];
    const float* vW1    = (const float*)d_in[10];
    const float* vb1    = (const float*)d_in[11];
    const float* vW2    = (const float*)d_in[12];
    const float* vb2    = (const float*)d_in[13];
    const float* combW  = (const float*)d_in[14];
    const float* combB  = (const float*)d_in[15];
    const float* Wout   = (const float*)d_in[16];
    float* out = (float*)d_out;

    // ---- workspace carve (256B aligned) ----
    size_t off = 0;
    auto alloc = [&](size_t bytes) -> char* {
        char* p = (char*)d_ws + off;
        off = align256(off + bytes);
        return p;
    };
    _Float16* xh      = (_Float16*)alloc((size_t)NSEQ * DIM * 2);
    _Float16* WqkvT   = (_Float16*)alloc((size_t)2048 * DIM * 2);
    _Float16* kW1T    = (_Float16*)alloc((size_t)HID * 2048 * 2);
    _Float16* vW1T    = (_Float16*)alloc((size_t)HID * 2048 * 2);
    _Float16* kW2T    = (_Float16*)alloc((size_t)DD * HID * 2);
    _Float16* vW2T    = (_Float16*)alloc((size_t)DD * HID * 2);
    _Float16* combWT  = (_Float16*)alloc((size_t)(3 * HH) * DIM * 2);
    _Float16* WoutT   = (_Float16*)alloc((size_t)DIM * DIM * 2);
    float*    qkv_f   = (float*)alloc((size_t)NSEQ * 2048 * 4);
    float*    q_f     = (float*)alloc((size_t)HH * NSEQ * DD * 4);
    float*    k_f     = (float*)alloc((size_t)KVH * NSEQ * DD * 4);
    float*    v_f     = (float*)alloc((size_t)KVH * NSEQ * DD * 4);
    _Float16* qh      = (_Float16*)alloc((size_t)HH * NSEQ * DD * 2);
    float*    rq_f    = (float*)alloc((size_t)HH * NSEQ * DD * 4);
    float*    rk_f    = (float*)alloc((size_t)KVH * NSEQ * DD * 4);
    _Float16* kwA     = (_Float16*)alloc((size_t)KVH * NCB * CBS * DD * 2);
    _Float16* vwA     = (_Float16*)alloc((size_t)KVH * NCB * CBS * DD * 2);
    _Float16* khid_h  = (_Float16*)alloc((size_t)KVH * NCB * HID * 2);
    _Float16* vhid_h  = (_Float16*)alloc((size_t)KVH * NCB * HID * 2);
    float*    ckbody  = (float*)alloc((size_t)KVH * NCB * DD * 4);
    float*    cvbody  = (float*)alloc((size_t)KVH * NCB * DD * 4);
    _Float16* ck_h    = (_Float16*)alloc((size_t)KVH * CK * DD * 2);
    float*    cv_f    = (float*)alloc((size_t)KVH * CK * DD * 4);
    _Float16* cvT_h   = (_Float16*)alloc((size_t)KVH * DD * CKP * 2);
    float*    csim_f  = (float*)alloc((size_t)HH * NSEQ * CK * 4);
    _Float16* cattn_h = (_Float16*)alloc((size_t)HH * NSEQ * CKP * 2);
    float*    cout_f  = (float*)alloc((size_t)HH * NSEQ * DD * 4);
    float*    fout_f  = (float*)alloc((size_t)HH * NSEQ * DD * 4);
    float*    sout_f  = (float*)alloc((size_t)HH * NSEQ * DD * 4);
    float*    gates_f = (float*)alloc((size_t)NSEQ * (3 * HH) * 4);
    int*      selidx  = (int*)alloc((size_t)KVH * NSEQ * NSEL * 4);
    int*      selok   = (int*)alloc((size_t)KVH * NSEQ * NSEL * 4);
    _Float16* comb_h  = (_Float16*)alloc((size_t)NSEQ * DIM * 2);

    // ---- 1. weight convert+transpose (f32 KxN -> f16 NxK) ----
    auto convT = [&](const float* W, _Float16* Wt, int K, int N) {
        const long long tot = (long long)K * N;
        convT_kernel<<<(unsigned)((tot + 255) / 256), 256, 0, stream>>>(W, Wt, K, N);
    };
    convT(Wqkv, WqkvT, DIM, 2048);
    convT(kW1, kW1T, HID, HID);
    convT(vW1, vW1T, HID, HID);
    convT(kW2, kW2T, HID, DD);
    convT(vW2, vW2T, HID, DD);
    convT(combW, combWT, DIM, 3 * HH);
    convT(Wout, WoutT, DIM, DIM);

    // ---- 2. RMSNorm ----
    rmsnorm_kernel<<<NSEQ, 256, 0, stream>>>(inp, norm_g, xh);

    // ---- 3. QKV projection (WMMA) ----
    launch_gemm(xh, WqkvT, qkv_f, nullptr, nullptr,
                NSEQ, 2048, DIM, 1.f, 0, 0, 0, 0, 1, 1, stream);

    // ---- 4. split ----
    split_qkv_kernel<<<(NSEQ * 2048) / 256, 256, 0, stream>>>(qkv_f, q_f, k_f, v_f, qh);

    // ---- 5. compression windows ----
    window_kernel<<<(KVH * NCB * CBS * DD) / 256, 256, 0, stream>>>(
        k_f, v_f, k_pos, v_pos, kwA, vwA);

    // ---- 6. compression MLPs (WMMA) ----
    launch_gemm(kwA, kW1T, nullptr, khid_h, kb1,
                KVH * NCB, HID, CBS * DD, 1.f, 1, 0, 0, 0, 1, 1, stream);
    launch_gemm(vwA, vW1T, nullptr, vhid_h, vb1,
                KVH * NCB, HID, CBS * DD, 1.f, 1, 0, 0, 0, 1, 1, stream);
    launch_gemm(khid_h, kW2T, ckbody, nullptr, kb2,
                KVH * NCB, DD, HID, 1.f, 0, 0, 0, 0, 1, 1, stream);
    launch_gemm(vhid_h, vW2T, cvbody, nullptr, vb2,
                KVH * NCB, DD, HID, 1.f, 0, 0, 0, 0, 1, 1, stream);

    // ---- 7. assemble ck / cv (with memory token), pad cvT ----
    hipMemsetAsync(cvT_h, 0, (size_t)KVH * DD * CKP * 2, stream);
    assemble_c_kernel<<<(KVH * CK * DD + 255) / 256, 256, 0, stream>>>(
        ckbody, cvbody, mem_kv, ck_h, cv_f, cvT_h);

    // ---- 8. compressed attention (WMMA, batched over 16 heads, GQA B-share) ----
    launch_gemm(qh, ck_h, csim_f, nullptr, nullptr,
                NSEQ, CK, DD, SCALE, 0,
                (long long)NSEQ * DD, (long long)CK * DD, (long long)NSEQ * CK,
                GQ, HH, stream);
    csoftmax_kernel<<<(HH * NSEQ) / 4, 128, 0, stream>>>(csim_f, cattn_h);
    launch_gemm(cattn_h, cvT_h, cout_f, nullptr, nullptr,
                NSEQ, DD, CKP, 1.f, 0,
                (long long)NSEQ * CKP, (long long)DD * CKP, (long long)NSEQ * DD,
                GQ, HH, stream);

    // ---- 9. RoPE ----
    rope_kernel<<<((HH + KVH) * NSEQ * (DD / 2)) / 256, 256, 0, stream>>>(
        q_f, k_f, rq_f, rk_f);

    // ---- 10. importance + top-k ----
    topk_kernel<<<(KVH * NSEQ) / 256, 256, 0, stream>>>(csim_f, selidx, selok);

    // ---- 11. fine attention ----
    fine_attn_kernel<<<(KVH * NSEQ) / 4, 128, 0, stream>>>(
        rq_f, rk_f, v_f, selidx, selok, fout_f);

    // ---- 12. sliding-window attention ----
    sliding_kernel<<<(HH * NSEQ) / 4, 128, 0, stream>>>(rq_f, rk_f, v_f, sout_f);

    // ---- 13. gates (WMMA) + combine ----
    launch_gemm(xh, combWT, gates_f, nullptr, combB,
                NSEQ, 3 * HH, DIM, 1.f, 0, 0, 0, 0, 1, 1, stream);
    combine_kernel<<<(NSEQ * DIM) / 256, 256, 0, stream>>>(
        gates_f, cout_f, fout_f, sout_f, comb_h);

    // ---- 14. output projection (WMMA) ----
    launch_gemm(comb_h, WoutT, out, nullptr, nullptr,
                NSEQ, DIM, DIM, 1.f, 0, 0, 0, 0, 1, 1, stream);
}
